// UFLAttention_19645180412783
// MI455X (gfx1250) — matrix operand order repaired and checked by reading, not compiled, not run on hardware
//
#include <hip/hip_runtime.h>

#define NH 8
#define NB 64
#define ND 512
#define EPSF 1e-8f
#define RSQRT_D 0.044194173824159216f   // 1/sqrt(512)
#define LOG2E   1.4426950408889634f

typedef __attribute__((ext_vector_type(16))) _Float16     v16h;
typedef __attribute__((ext_vector_type(8)))  float        v8f;
typedef __attribute__((ext_vector_type(4)))  unsigned int u32x4;
typedef __attribute__((ext_vector_type(8)))  int          i32x8;
typedef __attribute__((ext_vector_type(4)))  int          i32x4;

__global__ void ufl_init_out(const float* __restrict__ x, float* __restrict__ out) {
    int i = blockIdx.x * blockDim.x + threadIdx.x;
    if (i < NB * ND) out[i] = x[i];
}

#if __has_builtin(__builtin_amdgcn_tensor_load_to_lds)
#define HAVE_TDM 1
// Issue one TDM 1-D row load: 512 f32 from global -> LDS. D# per ISA ch.8.
// This toolchain exposes the 6-arg builtin: (g0, g1, g2, g3, g1ext, cpol).
__device__ __forceinline__ void tdm_load_row(const float* gptr, void* lptr) {
    unsigned long long ga = (unsigned long long)(uintptr_t)gptr;
    unsigned ldsoff = (unsigned)(uintptr_t)lptr;   // LDS aperture: addr[31:0] = offset
    u32x4 g0;
    g0.x = 1u;                                        // count=1, user descriptor
    g0.y = ldsoff;                                    // lds_addr (bits 63:32)
    g0.z = (unsigned)(ga & 0xffffffffu);              // global_addr[31:0]
    g0.w = (unsigned)((ga >> 32) & 0x1ffffffu)        // global_addr[56:32]
         | (2u << 30);                                // type = 2 ("image")
    i32x8 g1;
    g1[0] = (int)(2u << 16);                          // data_size = 4 bytes; wg_mask=0
    g1[1] = (int)((unsigned)(ND & 0xffff) << 16);     // tensor_dim0[15:0]=512
    g1[2] = (int)(1u << 16);                          // tensor_dim0 hi=0, tensor_dim1=1
    g1[3] = (int)((unsigned)ND << 16);                // tile_dim0 = 512
    g1[4] = 1;                                        // tile_dim1 = 1, tile_dim2 = 0
    g1[5] = ND;                                       // tensor_dim0_stride = 512
    g1[6] = 0;
    g1[7] = 0;
    i32x4 z4 = {0, 0, 0, 0};                          // groups 2/3 unused (<=2D tensor)
    i32x8 z8 = {0, 0, 0, 0, 0, 0, 0, 0};
    __builtin_amdgcn_tensor_load_to_lds(g0, g1, z4, z4, z8, 0);
}
#else
#define HAVE_TDM 0
#endif

// One workgroup per (h, b). 256 threads = 8 wave32s.
// Flash-attention-style two-pass softmax per 16-row i-tile; a single
// v_wmma_f32_16x16x32_f16 per K-chunk computes both sum(P*v) (col 0) and
// sum(P) (col 1) via B = [v | 1 | 0 ...] fetched straight from LDS in f16.
__launch_bounds__(256)
__global__ void ufl_attn_kernel(const float* __restrict__ x,
                                const float* __restrict__ Wq, const float* __restrict__ bq,
                                const float* __restrict__ Wk, const float* __restrict__ bk,
                                const float* __restrict__ Wv, const float* __restrict__ bv,
                                float* __restrict__ out) {
    __shared__ __align__(16) float    q_s[ND];
    __shared__ __align__(16) float    k_s[ND];
    __shared__ __align__(32) _Float16 vh_s[ND];      // v pre-converted to f16 (B source)
    __shared__ __align__(32) _Float16 onesh[16];     // B column 1
    __shared__ __align__(32) _Float16 zerosh[16];    // B columns 2..15
    __shared__ __align__(16) float    stage[4][ND];  // TDM dest: x, Wq, Wk, Wv rows
    __shared__ float wscr[8][16][2];                 // per-wave epilogue scratch

    const int h   = blockIdx.x % NH;
    const int bb  = blockIdx.x / NH;
    const int tid = threadIdx.x;

    __builtin_prefetch(&bq[h * ND], 0, 0);           // global_prefetch_b8
    __builtin_prefetch(&bk[h * ND], 0, 0);
    __builtin_prefetch(&bv[h * ND], 0, 0);

#if HAVE_TDM
    // Wave 0 DMAs the four input rows into LDS via the Tensor Data Mover.
    if (tid == 0) {
        tdm_load_row(&x [bb * ND], &stage[0][0]);
        tdm_load_row(&Wq[h  * ND], &stage[1][0]);
        tdm_load_row(&Wk[h  * ND], &stage[2][0]);
        tdm_load_row(&Wv[h  * ND], &stage[3][0]);
        __builtin_amdgcn_s_wait_tensorcnt(0);
    }
    __syncthreads();
    for (int d = tid; d < ND; d += 256) {
        float xv = stage[0][d];
        q_s[d]  = stage[1][d] * xv + bq[h * ND + d];
        k_s[d]  = stage[2][d] * xv + bk[h * ND + d];
        vh_s[d] = (_Float16)(stage[3][d] * xv + bv[h * ND + d]);
    }
#else
    for (int d = tid; d < ND; d += 256) {
        float xv = x[bb * ND + d];
        q_s[d]  = Wq[h * ND + d] * xv + bq[h * ND + d];
        k_s[d]  = Wk[h * ND + d] * xv + bk[h * ND + d];
        vh_s[d] = (_Float16)(Wv[h * ND + d] * xv + bv[h * ND + d]);
    }
#endif
    if (tid < 16) { onesh[tid] = (_Float16)1.0f; zerosh[tid] = (_Float16)0.0f; }
    __syncthreads();

    const int wave = tid >> 5;
    const int lane = tid & 31;
    const int ncol = lane & 15;   // A-row / B-column / D-column index
    const int hig  = lane >> 4;   // lane group (0: lanes 0-15, 1: lanes 16-31)

    // Per-lane B-fragment source: col 0 walks v (32B per chunk), col 1 = ones, rest = 0.
    const _Float16* bsrc = (ncol == 0) ? &vh_s[16 * hig]
                         : (ncol == 1) ? &onesh[0] : &zerosh[0];
    const int bstep = (ncol == 0) ? 32 : 0;   // halves per 32-wide K chunk

    for (int t = wave; t < ND / 16; t += 8) {
        const int i0 = t * 16;
        const float ki = k_s[i0 + ncol];   // this lane's row's k value

        // ---- Pass 1: row max of s = 1/(|q_j-k_i|+eps) == 1/(min_j |q_j-k_i| + eps).
        float dmin = 3.4e38f;
        const int jbeg = hig * (ND / 2);
        for (int j = jbeg; j < jbeg + ND / 2; j += 4) {
            float4 qv = *(const float4*)&q_s[j];
            dmin = fminf(dmin, fabsf(qv.x - ki));
            dmin = fminf(dmin, fabsf(qv.y - ki));
            dmin = fminf(dmin, fabsf(qv.z - ki));
            dmin = fminf(dmin, fabsf(qv.w - ki));
        }
        dmin = fminf(dmin, __shfl_xor(dmin, 16, 32));
        const float m = __builtin_amdgcn_rcpf(dmin + EPSF);
        const float nml2 = -m * LOG2E;     // exp(s-m) = exp2(s*log2e - m*log2e)

        // ---- Pass 2: accumulate C = P @ [v | 1 | 0...] over K-chunks of 32.
        v8f c = {};
        #pragma unroll 2
        for (int cidx = 0; cidx < ND / 32; ++cidx) {
            const int j0 = cidx * 32;
            // A fragment (16x32 f16), lane row = ncol:
            //   element e: K = e + 16*hig
            const int a0 = j0 + 16 * hig;
            float4 A0 = *(const float4*)&q_s[a0];
            float4 A1 = *(const float4*)&q_s[a0 + 4];
            float4 A2 = *(const float4*)&q_s[a0 + 8];
            float4 A3 = *(const float4*)&q_s[a0 + 12];
            float qa[16] = {A0.x, A0.y, A0.z, A0.w, A1.x, A1.y, A1.z, A1.w,
                            A2.x, A2.y, A2.z, A2.w, A3.x, A3.y, A3.z, A3.w};

            v16h afrag;
            #pragma unroll
            for (int e = 0; e < 16; ++e) {
                float s = __builtin_amdgcn_rcpf(fabsf(qa[e] - ki) + EPSF);
                afrag[e] = (_Float16)__builtin_amdgcn_exp2f(fmaf(s, LOG2E, nml2));
            }
            // B fragment: straight 32B LDS load, no VALU.
            v16h bfrag = *(const v16h*)(bsrc + cidx * bstep);

            c = __builtin_amdgcn_wmma_f32_16x16x32_f16(
                    /*neg_a=*/false, afrag, /*neg_b=*/false, bfrag,
                    /*c_mod=*/(short)0, c, /*reuse_a=*/false, /*reuse_b=*/false);
        }

        // ---- Epilogue: D[:,0] = sum_j P*v, D[:,1] = sum_j P (softmax denominator).
        // C/D layout: lane L, VGPR r -> (M = r + 8*(L>=16), N = L%16).
        if (ncol < 2) {
            #pragma unroll
            for (int r = 0; r < 8; ++r)
                wscr[wave][r + 8 * hig][ncol] = c[r];
        }
        asm volatile("s_wait_dscnt 0x0" ::: "memory");   // wave-local LDS RAW
        if (lane < 16) {
            float acc = wscr[wave][lane][0];
            float den = wscr[wave][lane][1];
            // v_rcp-based normalize (consistent with score math; avoids v_div_* chain)
            float res = acc * __builtin_amdgcn_rcpf(den) * RSQRT_D;
            atomicAdd(&out[bb * ND + i0 + lane], res);
        }
        asm volatile("s_wait_dscnt 0x0" ::: "memory");   // wscr reuse next tile
    }
}

extern "C" void kernel_launch(void* const* d_in, const int* in_sizes, int n_in,
                              void* d_out, int out_size, void* d_ws, size_t ws_size,
                              hipStream_t stream) {
    const float* x  = (const float*)d_in[0];
    const float* Wq = (const float*)d_in[1];
    const float* bq = (const float*)d_in[2];
    const float* Wk = (const float*)d_in[3];
    const float* bk = (const float*)d_in[4];
    const float* Wv = (const float*)d_in[5];
    const float* bv = (const float*)d_in[6];
    float* out = (float*)d_out;

    // out = x  (atomic accumulation target)
    ufl_init_out<<<(NB * ND + 255) / 256, 256, 0, stream>>>(x, out);
    // one block per (h, b); each block atomically adds its head's contribution
    ufl_attn_kernel<<<NH * NB, 256, 0, stream>>>(x, Wq, bq, Wk, bk, Wv, bv, out);
}
